// SRLEmbeddings_90726889161116
// MI455X (gfx1250) — matrix operand order repaired and checked by reading, not compiled, not run on hardware
//
#include <hip/hip_runtime.h>

// Reference shapes: B=16, S=16, L=128, D=768, A=8, T=8, PAD_ID=1.
// One workgroup per (b,s). Fused: one pass over emb -> avg + pred + arg0 + arg1.
// GEMM core per block: W[32x128] (f16, LDS) x emb[128x768] (f32 via TDM -> LDS),
// computed as 2 M-tiles x 48 N-tiles of v_wmma_f32_16x16x32_f16.

typedef _Float16 v8h  __attribute__((ext_vector_type(8)));
typedef _Float16 v16h __attribute__((ext_vector_type(16)));
typedef float    v8f  __attribute__((ext_vector_type(8)));
typedef unsigned int u32x4 __attribute__((ext_vector_type(4)));
typedef int          i32x4 __attribute__((ext_vector_type(4)));
typedef int          i32x8 __attribute__((ext_vector_type(8)));

#define L_DIM   128
#define D_DIM   768
#define T_DIM   8
#define PAD_ID  1
#define KC      32                    // K rows per chunk
#define HALF_N  384                   // columns per half tile
#define TILE_DW (KC * HALF_N)         // 12288 dwords per half tile
// TDM pad: +1 dword per 128 dwords -> padded row stride 387 dwords
#define PAD_ROW 387
#define TILE_PAD_DW (TILE_DW + TILE_DW / 128)   // 12384

#if __has_builtin(__builtin_amdgcn_tensor_load_to_lds)
#define HAVE_TDM 1
#else
#define HAVE_TDM 0
#endif

// LDS layout (bytes)
#define OFF_BUF0 0
#define OFF_BUF1 (TILE_PAD_DW * 4)                 // 49536
#define OFF_W    (TILE_PAD_DW * 8)                 // 99072
#define OFF_SID  (OFF_W + 32 * L_DIM * 2)          // +8KB
#define OFF_MASK (OFF_SID + L_DIM * 4)
#define OFF_TOK  (OFF_MASK + L_DIM * 4)
#define OFF_INVW (OFF_TOK + 192 * 4)
#define OFF_RNV  (OFF_INVW + 192 * 4)
#define SMEM_BYTES (OFF_RNV + 32 * 4)              // ~110KB

#if HAVE_TDM
// Issue one TDM load: 32 x 384 f32 tile (row stride 768 f32 in global) -> LDS,
// with +1 dword LDS padding every 128 dwords. Descriptor per ISA 8.3/8.4.
__device__ __forceinline__ void tdm_load_half(const float* gsrc, unsigned lds_off) {
    unsigned long long ga = (unsigned long long)(uintptr_t)gsrc;
    u32x4 g0 = { 1u,                                   // count=1, user mode
                 lds_off,                              // lds_addr
                 (unsigned)(ga & 0xffffffffu),         // global_addr[31:0]
                 (unsigned)((ga >> 32) & 0x1ffffffu) | (2u << 30) };  // [56:32]|type=2
    i32x8 g1 = { (int)((2u << 16) | (1u << 20) | (6u << 22)), // dsize=4B,pad,intvl=128dw
                 (int)((unsigned)D_DIM << 16),         // abar=0 | tensor_dim0.lo
                 (int)((unsigned)L_DIM << 16),         // tensor_dim0.hi=0 | tensor_dim1.lo
                 (int)((unsigned)HALF_N << 16),        // tensor_dim1.hi=0 | tile_dim0
                 (int)KC,                              // tile_dim1=32 | tile_dim2=0
                 (int)D_DIM,                           // tensor_dim0_stride.lo
                 0, 0 };
    i32x4 gz = {0, 0, 0, 0};
#if defined(__clang_major__) && __clang_major__ >= 23
    i32x8 gz8 = {0, 0, 0, 0, 0, 0, 0, 0};
    __builtin_amdgcn_tensor_load_to_lds(g0, g1, gz, gz, gz8, 0);
#else
    __builtin_amdgcn_tensor_load_to_lds(g0, g1, gz, gz, 0);
#endif
}
#else
// Fallback: synchronous cooperative copy with identical padded LDS layout.
__device__ __forceinline__ void coop_copy_half(float* buf, const float* gsrc, int tid) {
    for (int e = tid; e < TILE_DW; e += 256) {
        const int r = e / HALF_N, d = e - r * HALF_N;
        buf[e + (e >> 7)] = gsrc[(size_t)r * D_DIM + d];
    }
}
#endif

__global__ __launch_bounds__(256) void srl_fused_kernel(
    const int*   __restrict__ sid_g,    // [256,128]
    const int*   __restrict__ mask_g,   // [256,128]
    const float* __restrict__ emb_g,    // [256,128,768]
    const int*   __restrict__ pred_g,   // [256,64]
    const int*   __restrict__ a0_g,     // [256,64]
    const int*   __restrict__ a1_g,     // [256,64]
    float*       __restrict__ out)      // avg | pred | arg0 | arg1 (flat f32)
{
    extern __shared__ char smem[];
    float*    sBuf0 = (float*)(smem + OFF_BUF0);
    float*    sBuf1 = (float*)(smem + OFF_BUF1);
    _Float16* sW    = (_Float16*)(smem + OFF_W);
    int*      sSid  = (int*)(smem + OFF_SID);
    float*    sMask = (float*)(smem + OFF_MASK);
    int*      sTok  = (int*)(smem + OFF_TOK);
    float*    sInvw = (float*)(smem + OFF_INVW);
    float*    sRcpNV= (float*)(smem + OFF_RNV);

    const int bs  = blockIdx.x;
    const int tid = threadIdx.x;
    const float* embp = emb_g + (size_t)bs * L_DIM * D_DIM;

    // ---- kick off DMA for steps 0 and 1 immediately (overlaps W build) ---
#if HAVE_TDM
    const unsigned ldsBase = (unsigned)(uintptr_t)smem;
    if (tid < 32) {
        tdm_load_half(embp,          ldsBase + OFF_BUF0);
        tdm_load_half(embp + HALF_N, ldsBase + OFF_BUF1);
    }
#else
    coop_copy_half(sBuf0, embp,          tid);
    coop_copy_half(sBuf1, embp + HALF_N, tid);
#endif

    // ---- stage ids / masks / tokens -------------------------------------
    if (tid < L_DIM) {
        sSid[tid]  = sid_g[bs * L_DIM + tid];
        sMask[tid] = (float)mask_g[bs * L_DIM + tid];
    }
    if (tid < 192) {
        const int set = tid >> 6, i = tid & 63;
        const int* tg = (set == 0) ? pred_g : (set == 1) ? a0_g : a1_g;
        sTok[tid] = tg[bs * 64 + i];
    }
    __syncthreads();

    // ---- per-token 1/max(cnt,1) * valid ---------------------------------
    if (tid < 192) {
        const int tok = sTok[tid];
        int cnt = 0;
        #pragma unroll 4
        for (int l = 0; l < L_DIM; ++l) cnt += (sSid[l] == tok) ? 1 : 0;
        const float valid = (tok != PAD_ID) ? 1.0f : 0.0f;
        sInvw[tid] = valid / (float)(cnt > 1 ? cnt : 1);
    }
    __syncthreads();

    // ---- per-arg 1/max(n_valid,1), masked-mean 1/count ------------------
    if (tid < 24) {
        const int base = tid * T_DIM;
        int nv = 0;
        #pragma unroll
        for (int t = 0; t < T_DIM; ++t) nv += (sTok[base + t] != PAD_ID) ? 1 : 0;
        sRcpNV[tid] = 1.0f / (float)(nv > 1 ? nv : 1);
    } else if (tid == 24) {
        float c = 0.f;
        for (int l = 0; l < L_DIM; ++l) c += sMask[l];
        sRcpNV[24] = 1.0f / fmaxf(c, 1.0f);
    }
    __syncthreads();

    // ---- padded weight matrix W[32][128] f16 ----------------------------
    // rows 0-7 pred, 8-15 arg0, 16-23 arg1, 24 avg, 25-31 zero
    for (int e = tid; e < 32 * L_DIM; e += 256) {
        const int r = e >> 7, l = e & (L_DIM - 1);
        float w = 0.f;
        if (r < 24) {
            const int base = r * T_DIM;
            const int s = sSid[l];
            float acc = 0.f;
            #pragma unroll
            for (int t = 0; t < T_DIM; ++t)
                acc += (sTok[base + t] == s) ? sInvw[base + t] : 0.f;
            w = acc * sRcpNV[r];
        } else if (r == 24) {
            w = sMask[l] * sRcpNV[24];
        }
        sW[e] = (_Float16)w;
    }

    // ---- pipelined GEMM: 8 steps = 4 K-chunks x 2 column halves ---------
    const int wave   = tid >> 5;
    const int lane   = tid & 31;
    const int lane16 = lane & 15;
    const bool hi    = lane >= 16;
    const int kbase  = hi ? 16 : 0;

    v8f acc[6][2];
    const v8f z8 = {0.f, 0.f, 0.f, 0.f, 0.f, 0.f, 0.f, 0.f};
    #pragma unroll
    for (int a = 0; a < 6; ++a) { acc[a][0] = z8; acc[a][1] = z8; }

    #pragma unroll
    for (int s = 0; s < 8; ++s) {
        const int c = s >> 1, h = s & 1;
        float* buf = h ? sBuf1 : sBuf0;

#if HAVE_TDM
        if (tid < 32) {
            if (s < 7) __builtin_amdgcn_s_wait_tensorcnt(1);  // current step done
            else       __builtin_amdgcn_s_wait_tensorcnt(0);
        }
#endif
        __syncthreads();   // step-s tile visible to all waves

        // A-fragments for chunk c (contiguous k = 16*half + s, same as B)
        const int ko = hi ? 16 : 0;
        v16h afr[2];
        #pragma unroll
        for (int m = 0; m < 2; ++m) {
            const _Float16* wrow = &sW[(m * 16 + lane16) * L_DIM + c * KC];
            v8h lo = *(const v8h*)(wrow + ko);
            v8h h8 = *(const v8h*)(wrow + 8 + ko);
            afr[m] = __builtin_shufflevector(lo, h8, 0, 1, 2, 3, 4, 5, 6, 7,
                                                     8, 9, 10, 11, 12, 13, 14, 15);
        }

        // 3 N-tiles per wave per step; B gathered from padded f32 tile.
        // Padded addr is affine: (base + base>>7) + 387*j  (384 % 128 == 0)
        #pragma unroll
        for (int j3 = 0; j3 < 3; ++j3) {
            const int nloc = (wave * 3 + j3) * 16 + lane16;
            const int lin0 = kbase * HALF_N + nloc;
            const float* bp = buf + lin0 + (lin0 >> 7);
            v16h bfr;
            #pragma unroll
            for (int j = 0; j < 16; ++j)
                bfr[j] = (_Float16)bp[PAD_ROW * j];

            const int a = h * 3 + j3;
            acc[a][0] = __builtin_amdgcn_wmma_f32_16x16x32_f16(
                false, afr[0], false, bfr, (short)0, acc[a][0], false, false);
            acc[a][1] = __builtin_amdgcn_wmma_f32_16x16x32_f16(
                false, afr[1], false, bfr, (short)0, acc[a][1], false, false);
        }

        __syncthreads();   // everyone done reading buf before it is refilled

        if (s + 2 < 8) {
            const int c2 = (s + 2) >> 1, h2 = (s + 2) & 1;  // h2 == h
            const float* gsrc = embp + (size_t)c2 * KC * D_DIM + h2 * HALF_N;
#if HAVE_TDM
            if (tid < 32)
                tdm_load_half(gsrc, ldsBase + (h2 ? OFF_BUF1 : OFF_BUF0));
#else
            coop_copy_half(h2 ? sBuf1 : sBuf0, gsrc, tid);
#endif
        }
    }

    // ---- scatter C/D tiles to the four concatenated outputs -------------
    // C/D layout: VGPR v, lanes 0-15 -> M=v, lanes 16-31 -> M=v+8
    const size_t OFF_PRED = (size_t)256 * D_DIM;
    const size_t OFF_A0   = OFF_PRED + (size_t)256 * 8 * D_DIM;
    const size_t OFF_A1   = OFF_A0   + (size_t)256 * 8 * D_DIM;

    #pragma unroll
    for (int a = 0; a < 6; ++a) {
        const int nt  = (a >= 3 ? 24 : 0) + wave * 3 + (a % 3);
        const int col = nt * 16 + lane16;
        #pragma unroll
        for (int v = 0; v < 8; ++v) {
            const size_t rowoff = ((size_t)(bs * 8 + v)) * D_DIM + col;
            // M-tile 0: rows 0-7 predicate (lo lanes), rows 8-15 arg0 (hi lanes)
            out[(hi ? OFF_A0 : OFF_PRED) + rowoff] = acc[a][0][v];
            // M-tile 1: rows 16-23 arg1 (lo lanes); row 24 avg (hi lanes, v==0)
            if (!hi)          out[OFF_A1 + rowoff] = acc[a][1][v];
            else if (v == 0)  out[(size_t)bs * D_DIM + col] = acc[a][1][v];
        }
    }
}

extern "C" void kernel_launch(void* const* d_in, const int* in_sizes, int n_in,
                              void* d_out, int out_size, void* d_ws, size_t ws_size,
                              hipStream_t stream) {
    (void)in_sizes; (void)n_in; (void)out_size; (void)d_ws; (void)ws_size;
    const int*   sid   = (const int*)d_in[0];
    const int*   mask  = (const int*)d_in[1];
    const float* emb   = (const float*)d_in[2];
    const int*   predi = (const int*)d_in[3];
    const int*   a0    = (const int*)d_in[4];
    const int*   a1    = (const int*)d_in[5];
    float*       out   = (float*)d_out;

    srl_fused_kernel<<<dim3(256), dim3(256), SMEM_BYTES, stream>>>(
        sid, mask, emb, predi, a0, a1, out);
}
